// Block_26843545600813
// MI455X (gfx1250) — matrix operand order repaired and checked by reading, not compiled, not run on hardware
//
#include <hip/hip_runtime.h>

// ---------------------------------------------------------------------------
// Types
// ---------------------------------------------------------------------------
typedef __bf16 bf16;
typedef __attribute__((ext_vector_type(8)))  bf16  bf16x8;
typedef __attribute__((ext_vector_type(16))) bf16  bf16x16;
typedef __attribute__((ext_vector_type(8)))  float f32x8;

// Problem constants (match reference)
constexpr int Bb = 4, Ss = 1024, Dd = 768, Hh = 12, DFf = 4 * Dd;
constexpr int BS = Bb * Ss;            // 4096 total rows
constexpr int HDH = Hh * Dd;           // 9216 concat width

// ---------------------------------------------------------------------------
// WMMA helpers (CDNA5 bf16 16x16x32, f32 accumulate)
// ---------------------------------------------------------------------------
static __device__ __forceinline__ bf16x16 ldfrag(const bf16* p0, const bf16* p1) {
  bf16x8 lo = *(const bf16x8*)p0;
  bf16x8 hi = *(const bf16x8*)p1;
  return __builtin_shufflevector(lo, hi, 0,1,2,3,4,5,6,7,8,9,10,11,12,13,14,15);
}
static __device__ __forceinline__ f32x8 wmma_bf16(bf16x16 a, bf16x16 b, f32x8 c) {
  return __builtin_amdgcn_wmma_f32_16x16x32_bf16(false, a, false, b, (short)0, c,
                                                 false, false);
}

// ---------------------------------------------------------------------------
// CDNA5 async global->LDS copy (no VGPR staging, tracked by ASYNCcnt).
// LDS address = low 32 bits of the generic pointer (ISA 10.2 truncation).
// ---------------------------------------------------------------------------
static __device__ __forceinline__ void async_g2l_b128(void* ldsp, const void* gp) {
  unsigned loff = (unsigned)(unsigned long long)(__UINTPTR_TYPE__)ldsp;
  unsigned long long ga = (unsigned long long)(__UINTPTR_TYPE__)gp;
  asm volatile("global_load_async_to_lds_b128 %0, %1, off"
               :: "v"(loff), "v"(ga) : "memory");
}
static __device__ __forceinline__ void wait_async() {
  asm volatile("s_wait_asynccnt 0x0" ::: "memory");
}

// ---------------------------------------------------------------------------
// LayerNorm (f32 in -> bf16 out), one block per row of 768
// ---------------------------------------------------------------------------
__global__ __launch_bounds__(256) void layernorm_bf16(
    const float* __restrict__ x, const float* __restrict__ g,
    const float* __restrict__ be, bf16* __restrict__ outp) {
  __shared__ float r1[256], r2[256];
  __shared__ float mu_s, rstd_s;
  const int row = blockIdx.x, tid = threadIdx.x;
  const float* xr = x + (long long)row * Dd;
  float v[3]; float s = 0.f, ss = 0.f;
#pragma unroll
  for (int i = 0; i < 3; ++i) { v[i] = xr[tid + i * 256]; s += v[i]; ss += v[i] * v[i]; }
  r1[tid] = s; r2[tid] = ss; __syncthreads();
  for (int off = 128; off > 0; off >>= 1) {
    if (tid < off) { r1[tid] += r1[tid + off]; r2[tid] += r2[tid + off]; }
    __syncthreads();
  }
  if (tid == 0) {
    float mu = r1[0] * (1.f / Dd);
    float var = r2[0] * (1.f / Dd) - mu * mu;
    mu_s = mu; rstd_s = rsqrtf(var + 1e-5f);
  }
  __syncthreads();
  const float mu = mu_s, rstd = rstd_s;
#pragma unroll
  for (int i = 0; i < 3; ++i) {
    int c = tid + i * 256;
    outp[(long long)row * Dd + c] = (bf16)((v[i] - mu) * rstd * g[c] + be[c]);
  }
}

// ---------------------------------------------------------------------------
// Transpose + convert f32[R][C] -> bf16[C][R] (per batch-z), 32x32 tiles
// ---------------------------------------------------------------------------
__global__ __launch_bounds__(256) void transpose_to_bf16(
    const float* __restrict__ in, bf16* __restrict__ outp, int R, int C) {
  __shared__ float t[32][33];
  const long long base = (long long)blockIdx.z * R * C;
  const int r0 = blockIdx.y * 32, c0 = blockIdx.x * 32;
  const int tx = threadIdx.x & 31, ty = threadIdx.x >> 5;
#pragma unroll
  for (int i = 0; i < 4; ++i)
    t[ty + i * 8][tx] = in[base + (long long)(r0 + ty + i * 8) * C + c0 + tx];
  __syncthreads();
#pragma unroll
  for (int i = 0; i < 4; ++i)
    outp[base + (long long)(c0 + ty + i * 8) * R + r0 + tx] = (bf16)t[tx][ty + i * 8];
}

// ---------------------------------------------------------------------------
// Batched WMMA GEMM: C[M,N] = A[M,K] (row-major bf16) * Bt[N,K] (n-major bf16)
// 128x128x32 tiles, 8 waves (2x4), 8 wmma tiles per wave per k-step.
// Double-buffered LDS filled by async global->LDS copies (ASYNCcnt).
// Epilogue: +bias[n], optional ReLU, optional +resid (f32, row-major),
// bf16 or f32 store, optional transposed store (out[n*M+m]).
// ---------------------------------------------------------------------------
constexpr int BM = 128, BN = 128, BK = 32, LDT = BK + 8;  // padded LDS stride

template <bool TRANS_OUT, bool RELU, bool RESID, bool OUT_BF16>
__global__ __launch_bounds__(256) void gemm_wmma(
    const bf16* __restrict__ A, const bf16* __restrict__ Bt,
    const float* __restrict__ bias, const float* __restrict__ resid,
    void* __restrict__ Cout, int M, int N, int K,
    int aDiv, long long aStride, int bMod, long long bStride,
    long long cStride, long long biasStride) {
  __shared__ bf16 As[2][BM * LDT];
  __shared__ bf16 Bs[2][BN * LDT];
  const int tid = threadIdx.x;
  const int wave = tid >> 5, lane = tid & 31, hf = lane >> 4, ln = lane & 15;
  const int wm = wave & 1, wn = wave >> 1;
  const int z = blockIdx.z;
  const long long aOff = (long long)(z / aDiv) * aStride;
  const long long bOff = (long long)(z % bMod) * bStride;
  const long long cOff = (long long)z * cStride;
  const long long biasOff = (long long)(z % bMod) * biasStride;
  const int bm0 = blockIdx.y * BM, bn0 = blockIdx.x * BN;

  const int lrow = tid >> 2, lc8 = (tid & 3) * 8;  // this thread's tile slots
  auto issue_tiles = [&](int k0, int buf) {
#pragma unroll
    for (int i = 0; i < 2; ++i) {
      int row = lrow + i * 64;
      async_g2l_b128(&As[buf][row * LDT + lc8],
                     &A[aOff + (long long)(bm0 + row) * K + k0 + lc8]);
      async_g2l_b128(&Bs[buf][row * LDT + lc8],
                     &Bt[bOff + (long long)(bn0 + row) * K + k0 + lc8]);
    }
  };

  f32x8 acc[4][2] = {};

  issue_tiles(0, 0);
  wait_async();
  __syncthreads();
  int cur = 0;

  for (int k0 = 0; k0 < K; k0 += BK) {
    const int nxt = cur ^ 1;
    if (k0 + BK < K) issue_tiles(k0 + BK, nxt);  // overlap with WMMA below

    bf16x16 af[4], bfv[2];
#pragma unroll
    for (int tm = 0; tm < 4; ++tm) {
      const bf16* p = &As[cur][(wm * 64 + tm * 16 + ln) * LDT];
      af[tm] = ldfrag(p + hf * 8, p + 16 + hf * 8);
    }
#pragma unroll
    for (int tn = 0; tn < 2; ++tn) {
      const bf16* p = &Bs[cur][(wn * 32 + tn * 16 + ln) * LDT];
      bfv[tn] = ldfrag(p + hf * 8, p + 16 + hf * 8);
    }
#pragma unroll
    for (int tm = 0; tm < 4; ++tm)
#pragma unroll
      for (int tn = 0; tn < 2; ++tn)
        acc[tm][tn] = wmma_bf16(af[tm], bfv[tn], acc[tm][tn]);

    wait_async();
    __syncthreads();
    cur = nxt;
  }

  // Epilogue. C-frag layout: m = r + 8*half, n = lane&15.
#pragma unroll
  for (int tm = 0; tm < 4; ++tm) {
#pragma unroll
    for (int tn = 0; tn < 2; ++tn) {
      const int col = bn0 + wn * 32 + tn * 16 + ln;
      const float bv = bias[biasOff + col];
#pragma unroll
      for (int r = 0; r < 8; ++r) {
        const int row = bm0 + wm * 64 + tm * 16 + r + hf * 8;
        float v = acc[tm][tn][r] + bv;
        if (RELU) v = fmaxf(v, 0.f);
        if (RESID) v += resid[(long long)row * N + col];
        const long long o =
            cOff + (TRANS_OUT ? (long long)col * M + row : (long long)row * N + col);
        if (OUT_BF16) ((bf16*)Cout)[o] = (bf16)v;
        else          ((float*)Cout)[o] = v;
      }
    }
  }
}

// ---------------------------------------------------------------------------
// Flash-style attention. grid = (S/32, B*H), 256 threads.
// Q:[bh][S][DH] bf16, K:[bh][S][DH] bf16, VT:[bh][DH][S] bf16.
// Out: concat [B][S][H*DH] bf16.
// ---------------------------------------------------------------------------
constexpr int QT = 32, KT = 64;
constexpr int LQ = Dd + 8, LK = 40, LS = 68, LP = 72, LV = 40;
constexpr int LKBUF = KT * LK;  // one K-chunk buffer (double-buffered)

__global__ __launch_bounds__(256) void attn_flash(
    const bf16* __restrict__ Q, const bf16* __restrict__ Kmat,
    const bf16* __restrict__ VT, bf16* __restrict__ Outc) {
  extern __shared__ char smem[];
  bf16*  lq    = (bf16*)smem;                 // [QT][LQ]     persistent Q tile
  bf16*  lk    = lq + QT * LQ;                // [2][KT][LK]  K chunk (dbl-buffered)
  bf16*  lp    = lk + 2 * LKBUF;              // [QT][LP]     P tile bf16
  float* ls    = (float*)(lp + QT * LP);      // [QT][LS]     scores f32
  bf16*  lv    = (bf16*)(ls + QT * LS);       // [Dd][LV]     V^T chunk
  float* lstat = (float*)(lv + Dd * LV);      // m[32], l[32], scale[32]
  float* lred  = lstat + 96;                  // [32][8] reduction scratch

  const int tid = threadIdx.x;
  const int wave = tid >> 5, lane = tid & 31, hf = lane >> 4, ln = lane & 15;
  const int bh = blockIdx.y, bb = bh / Hh, hq = bh - bb * Hh;
  const int s0 = blockIdx.x * QT;
  const bf16* Qb = Q    + (long long)bh * Ss * Dd;
  const bf16* Kb = Kmat + (long long)bh * Ss * Dd;
  const bf16* Vb = VT   + (long long)bh * Dd * Ss;
  const float scale = 0.03608439182435161f;  // 768^-0.5

  // Async-load full Q tile (32 x 768) into LDS once.
#pragma unroll
  for (int i = 0; i < 12; ++i) {
    int idx = tid + i * 256;                 // 0..3071
    int row = idx / 96, c8 = (idx % 96) * 8;
    async_g2l_b128(&lq[row * LQ + c8], &Qb[(long long)(s0 + row) * Dd + c8]);
  }
  if (tid < 32) { lstat[tid] = -1e30f; lstat[32 + tid] = 0.f; }

  f32x8 ctx[2][6] = {};                      // 32 rows x 96 cols of DH per wave
  const int sam = wave & 1, san = wave >> 1; // stage-A 16x16 tile coords (2x4)
  const int e0 = wave * 96;                  // stage-C DH slice base
  const int krow = tid >> 2, kc8 = (tid & 3) * 8;  // K-chunk load slot
  wait_async();
  __syncthreads();

  for (int t0 = 0; t0 < Ss; t0 += KT) {
    // ---- Stage A: S[32x64] = Q_tile . K_tile^T (gemm-K = DH) ----
    async_g2l_b128(&lk[krow * LK + kc8], &Kb[(long long)(t0 + krow) * Dd + kc8]);
    wait_async();
    __syncthreads();
    int cur = 0;
    f32x8 sacc = {};
    for (int kk = 0; kk < Dd; kk += BK) {
      const int nxt = cur ^ 1;
      if (kk + BK < Dd)
        async_g2l_b128(&lk[nxt * LKBUF + krow * LK + kc8],
                       &Kb[(long long)(t0 + krow) * Dd + kk + BK + kc8]);
      const bf16* pa = &lq[(sam * 16 + ln) * LQ + kk];
      bf16x16 a = ldfrag(pa + hf * 8, pa + 16 + hf * 8);
      const bf16* pb = &lk[cur * LKBUF + (san * 16 + ln) * LK];
      bf16x16 b = ldfrag(pb + hf * 8, pb + 16 + hf * 8);
      sacc = wmma_bf16(a, b, sacc);
      wait_async();
      __syncthreads();
      cur = nxt;
    }
#pragma unroll
    for (int r = 0; r < 8; ++r)
      ls[(sam * 16 + r + hf * 8) * LS + san * 16 + ln] = sacc[r];
    __syncthreads();

    // ---- Stage B: online softmax (8 threads per row) ----
    const int rrow = tid >> 3, sub = tid & 7;
    float pm = -1e30f;
#pragma unroll
    for (int j = 0; j < 8; ++j) pm = fmaxf(pm, ls[rrow * LS + sub * 8 + j] * scale);
    lred[rrow * 8 + sub] = pm;
    __syncthreads();
    if (tid < 32) {
      float mo = lstat[tid], mc = lred[tid * 8];
      for (int j = 1; j < 8; ++j) mc = fmaxf(mc, lred[tid * 8 + j]);
      float mn = fmaxf(mo, mc);
      lstat[tid] = mn;
      lstat[64 + tid] = __expf(mo - mn);     // ctx rescale factor
    }
    __syncthreads();
    const float mn = lstat[rrow];
    float psum = 0.f;
#pragma unroll
    for (int j = 0; j < 8; ++j) {
      float p = __expf(ls[rrow * LS + sub * 8 + j] * scale - mn);
      lp[rrow * LP + sub * 8 + j] = (bf16)p;
      psum += p;
    }
    lred[rrow * 8 + sub] = psum;
    __syncthreads();
    if (tid < 32) {
      float s = 0.f;
      for (int j = 0; j < 8; ++j) s += lred[tid * 8 + j];
      lstat[32 + tid] = lstat[32 + tid] * lstat[64 + tid] + s;
    }
    __syncthreads();

    // rescale running ctx by exp(m_old - m_new)
#pragma unroll
    for (int tmi = 0; tmi < 2; ++tmi)
#pragma unroll
      for (int r = 0; r < 8; ++r) {
        const float sc = lstat[64 + tmi * 16 + r + hf * 8];
#pragma unroll
        for (int tni = 0; tni < 6; ++tni) ctx[tmi][tni][r] *= sc;
      }

    // ---- Stage C: ctx += P[32x64] . V[64x768] (two k-chunks of 32) ----
    for (int tt = 0; tt < 2; ++tt) {
      __syncthreads();   // previous chunk's readers done before overwrite
#pragma unroll
      for (int i = 0; i < 12; ++i) {
        int idx = tid + i * 256;             // 0..3071
        int row = idx >> 2, c8 = (idx & 3) * 8;
        async_g2l_b128(&lv[row * LV + c8],
                       &Vb[(long long)row * Ss + t0 + tt * 32 + c8]);
      }
      wait_async();
      __syncthreads();
      bf16x16 pa[2];
#pragma unroll
      for (int tmi = 0; tmi < 2; ++tmi) {
        const bf16* p = &lp[(tmi * 16 + ln) * LP + tt * 32];
        pa[tmi] = ldfrag(p + hf * 8, p + 16 + hf * 8);
      }
#pragma unroll
      for (int tni = 0; tni < 6; ++tni) {
        const bf16* p = &lv[(e0 + tni * 16 + ln) * LV];
        bf16x16 bv = ldfrag(p + hf * 8, p + 16 + hf * 8);
#pragma unroll
        for (int tmi = 0; tmi < 2; ++tmi)
          ctx[tmi][tni] = wmma_bf16(pa[tmi], bv, ctx[tmi][tni]);
      }
    }
    __syncthreads();
  }

  // ---- Finalize: divide by row sum, write concat[b, s, h*DH + e] ----
#pragma unroll
  for (int tmi = 0; tmi < 2; ++tmi)
#pragma unroll
    for (int r = 0; r < 8; ++r) {
      const int row = tmi * 16 + r + hf * 8;
      const float rl = 1.f / lstat[32 + row];
#pragma unroll
      for (int tni = 0; tni < 6; ++tni) {
        const int e = e0 + tni * 16 + ln;
        const long long o =
            ((long long)bb * Ss + s0 + row) * (long long)HDH + (long long)hq * Dd + e;
        Outc[o] = (bf16)(ctx[tmi][tni][r] * rl);
      }
    }
}

// ---------------------------------------------------------------------------
// Host launcher
// ---------------------------------------------------------------------------
extern "C" void kernel_launch(void* const* d_in, const int* in_sizes, int n_in,
                              void* d_out, int out_size, void* d_ws, size_t ws_size,
                              hipStream_t stream) {
  const float* x   = (const float*)d_in[0];
  const float* Wq  = (const float*)d_in[1];
  const float* bq  = (const float*)d_in[2];
  const float* Wk  = (const float*)d_in[3];
  const float* bk  = (const float*)d_in[4];
  const float* Wv  = (const float*)d_in[5];
  const float* bv  = (const float*)d_in[6];
  const float* Wo  = (const float*)d_in[7];
  const float* bo  = (const float*)d_in[8];
  const float* W1  = (const float*)d_in[9];
  const float* b1  = (const float*)d_in[10];
  const float* W2  = (const float*)d_in[11];
  const float* b2  = (const float*)d_in[12];
  const float* g1  = (const float*)d_in[13];
  const float* be1 = (const float*)d_in[14];
  const float* g2  = (const float*)d_in[15];
  const float* be2 = (const float*)d_in[16];
  float* out = (float*)d_out;

  char* wsp = (char*)d_ws;
  auto carve = [&](size_t bytes) -> char* {
    char* p = wsp; wsp += (bytes + 255) & ~(size_t)255; return p;
  };
  bf16* hb    = (bf16*)carve((size_t)BS * Dd * 2);
  bf16* WqT   = (bf16*)carve((size_t)Hh * Dd * Dd * 2);
  bf16* WkT   = (bf16*)carve((size_t)Hh * Dd * Dd * 2);
  bf16* WvT   = (bf16*)carve((size_t)Hh * Dd * Dd * 2);
  bf16* WoT   = (bf16*)carve((size_t)HDH * Dd * 2);
  bf16* W1T   = (bf16*)carve((size_t)Dd * DFf * 2);
  bf16* W2T   = (bf16*)carve((size_t)DFf * Dd * 2);
  bf16* Qbuf  = (bf16*)carve((size_t)Bb * Hh * Ss * Dd * 2);
  bf16* Kbuf  = (bf16*)carve((size_t)Bb * Hh * Ss * Dd * 2);
  bf16* VTbuf = (bf16*)carve((size_t)Bb * Hh * Dd * Ss * 2);
  bf16* concat= (bf16*)carve((size_t)BS * HDH * 2);
  float* x1b  = (float*)carve((size_t)BS * Dd * 4);
  bf16* h2b   = (bf16*)carve((size_t)BS * Dd * 2);
  bf16* ffn1  = (bf16*)carve((size_t)BS * DFf * 2);

  // 1) LN1 -> bf16 h
  layernorm_bf16<<<BS, 256, 0, stream>>>(x, g1, be1, hb);

  // 2) weight transpose/convert to n-major bf16
  transpose_to_bf16<<<dim3(24, 24, Hh),  256, 0, stream>>>(Wq, WqT, Dd, Dd);
  transpose_to_bf16<<<dim3(24, 24, Hh),  256, 0, stream>>>(Wk, WkT, Dd, Dd);
  transpose_to_bf16<<<dim3(24, 24, Hh),  256, 0, stream>>>(Wv, WvT, Dd, Dd);
  transpose_to_bf16<<<dim3(24, 288, 1),  256, 0, stream>>>(Wo, WoT, HDH, Dd);
  transpose_to_bf16<<<dim3(96, 24, 1),   256, 0, stream>>>(W1, W1T, Dd, DFf);
  transpose_to_bf16<<<dim3(24, 96, 1),   256, 0, stream>>>(W2, W2T, DFf, Dd);

  // 3) QKV projections (batched over B*H); V stored transposed [DH][S]
  const long long aS = (long long)Ss * Dd, bSd = (long long)Dd * Dd,
                  cS = (long long)Ss * Dd;
  gemm_wmma<false,false,false,true><<<dim3(6, 8, Bb * Hh), 256, 0, stream>>>(
      hb, WqT, bq, nullptr, Qbuf, Ss, Dd, Dd, Hh, aS, Hh, bSd, cS, Dd);
  gemm_wmma<false,false,false,true><<<dim3(6, 8, Bb * Hh), 256, 0, stream>>>(
      hb, WkT, bk, nullptr, Kbuf, Ss, Dd, Dd, Hh, aS, Hh, bSd, cS, Dd);
  gemm_wmma<true,false,false,true><<<dim3(6, 8, Bb * Hh), 256, 0, stream>>>(
      hb, WvT, bv, nullptr, VTbuf, Ss, Dd, Dd, Hh, aS, Hh, bSd, cS, Dd);

  // 4) flash attention -> concat (bf16)
  const int smem_attn =
      (QT * LQ + 2 * LKBUF + QT * LP + Dd * LV) * 2 + (QT * LS + 96 + 256) * 4;
  (void)hipFuncSetAttribute((const void*)attn_flash,
                            hipFuncAttributeMaxDynamicSharedMemorySize, smem_attn);
  attn_flash<<<dim3(Ss / QT, Bb * Hh), 256, smem_attn, stream>>>(
      Qbuf, Kbuf, VTbuf, concat);

  // 5) output projection + residual -> x1 (f32)
  gemm_wmma<false,false,true,false><<<dim3(6, 32, 1), 256, 0, stream>>>(
      concat, WoT, bo, x, x1b, BS, Dd, HDH, 1, 0, 1, 0, 0, 0);

  // 6) LN2 -> bf16 h2
  layernorm_bf16<<<BS, 256, 0, stream>>>(x1b, g2, be2, h2b);

  // 7) FFN1 (ReLU, bf16)
  gemm_wmma<false,true,false,true><<<dim3(24, 32, 1), 256, 0, stream>>>(
      h2b, W1T, b1, nullptr, ffn1, BS, DFf, Dd, 1, 0, 1, 0, 0, 0);

  // 8) FFN2 + residual x1 -> d_out (f32)
  gemm_wmma<false,false,true,false><<<dim3(6, 32, 1), 256, 0, stream>>>(
      ffn1, W2T, b2, x1b, out, BS, Dd, DFf, 1, 0, 1, 0, 0, 0);
}
